// Featup_without_global_48326972014843
// MI455X (gfx1250) — compile-verified, matrix operand order checked by reading, not run on hardware
//
#include <hip/hip_runtime.h>

// ---------------- problem constants ----------------
#define BATCH   8
#define CCH     384
#define HH      224
#define WW      224
#define HW      (HH*WW)
#define NPTS    80
#define NTOK    240          // 3*NPTS tokens
#define DMODEL  386
#define DPAD    416          // 13 * 32  (WMMA K-steps of 32)
#define FFDIM   2048
#define ATTPAD  256          // 240 padded to multiple of 32 for A*V GEMM

typedef __attribute__((ext_vector_type(16))) __bf16 v16bf;
typedef __attribute__((ext_vector_type(8)))  __bf16 v8bf;
typedef __attribute__((ext_vector_type(8)))  float  v8f;

__device__ __forceinline__ __bf16 f2bf(float f) {
    union { float f; unsigned u; } in; in.f = f;
    unsigned u = in.u;
    unsigned r = u + 0x7FFFu + ((u >> 16) & 1u);   // round-to-nearest-even
    union { unsigned short s; __bf16 b; } out;
    out.s = (unsigned short)(r >> 16);
    return out.b;
}

// Load one 16x32 bf16 fragment row-chunk per the CDNA5 A/B layout:
// lanes 0-15 hold K = k0+0..7 and k0+16..23 ; lanes 16-31 hold K = k0+8..15 and k0+24..31
__device__ __forceinline__ v16bf ldfrag(const __bf16* __restrict__ row, int k0, int h) {
    const v8bf lo = *reinterpret_cast<const v8bf*>(row + k0 + h * 8);
    const v8bf hi = *reinterpret_cast<const v8bf*>(row + k0 + 16 + h * 8);
    v16bf o;
#pragma unroll
    for (int e = 0; e < 8; ++e) { o[e] = lo[e]; o[8 + e] = hi[e]; }
    return o;
}

// ---------------- generic bf16 WMMA GEMM:  C = A(MxK) * B(NxK)^T  ----------------
struct GemmArgs {
    const __bf16* A;  int lda;  long long sA;   // per-batch stride (elements)
    const __bf16* B;  int ldb;  long long sB;   // 0 => shared weights
    const float*  bias; int biasN;
    float scale; int relu;
    float*  Cf; int ldcf; long long sCf;        // optional f32 output
    __bf16* Cb; int ldcb; long long sCb;        // optional bf16 output
    int transCb;                                 // store Cb transposed (n-major)
    int ksteps;                                  // K / 32
};

__global__ void gemm_bf16_wmma(GemmArgs g) {
    const int nt = blockIdx.x, mt = blockIdx.y, b = blockIdx.z;
    const int l = threadIdx.x;
    const int r = l & 15;        // row within 16-tile
    const int h = l >> 4;        // half-wave selector
    const __bf16* Arow = g.A + (long long)b * g.sA + (long long)(mt * 16 + r) * g.lda;
    const __bf16* Brow = g.B + (long long)b * g.sB + (long long)(nt * 16 + r) * g.ldb;
    v8f c = {};
    for (int kt = 0; kt < g.ksteps; ++kt) {
        const int k0 = kt * 32;
        v16bf a  = ldfrag(Arow, k0, h);
        v16bf bb = ldfrag(Brow, k0, h);
        c = __builtin_amdgcn_wmma_f32_16x16x32_bf16(false, a, false, bb,
                                                    (short)0, c, false, false);
    }
    const int nn = nt * 16 + r;
    const float bv = (g.bias && nn < g.biasN) ? g.bias[nn] : 0.0f;
#pragma unroll
    for (int e = 0; e < 8; ++e) {
        const int m = mt * 16 + e + h * 8;     // C/D layout: VGPR e, lanes 0-15 -> M=e ; 16-31 -> M=e+8
        float v = (c[e] + bv) * g.scale;
        if (g.relu) v = fmaxf(v, 0.0f);
        if (g.Cf) g.Cf[(long long)b * g.sCf + (long long)m * g.ldcf + nn] = v;
        if (g.Cb) {
            long long idx = g.transCb
                ? ((long long)b * g.sCb + (long long)nn * g.ldcb + m)
                : ((long long)b * g.sCb + (long long)m * g.ldcb + nn);
            g.Cb[idx] = f2bf(v);
        }
    }
}

// ---------------- weight pad/convert: src(rows x cols f32) -> dst(npad x kpad bf16, zero pad) ----------------
__global__ void pad_w_kernel(const float* __restrict__ src, __bf16* __restrict__ dst,
                             int rows, int cols, int kpad, long long total) {
    long long i = (long long)blockIdx.x * blockDim.x + threadIdx.x;
    if (i >= total) return;
    int n = (int)(i / kpad), k = (int)(i % kpad);
    float v = (n < rows && k < cols) ? src[(long long)n * cols + k] : 0.0f;
    dst[i] = f2bf(v);
}

// ---------------- boundary alignment (roll-argmin over 80 shifts) ----------------
__global__ void align_kernel(const int* __restrict__ firstB, const int* __restrict__ prevB,
                             int* __restrict__ firstOut) {
    const int b = blockIdx.x, tid = threadIdx.x;
    __shared__ int dist[NPTS];
    __shared__ int best;
    const int* fb = firstB + b * NPTS * 2;
    const int* pb = prevB  + b * NPTS * 2;
    for (int r = tid; r < NPTS; r += blockDim.x) {
        int s = 0;
        for (int j = 0; j < NPTS; ++j) {
            int jj = j - r; if (jj < 0) jj += NPTS;
            s += abs(pb[j * 2] - fb[jj * 2]) + abs(pb[j * 2 + 1] - fb[jj * 2 + 1]);
        }
        dist[r] = s;
    }
    __syncthreads();
    if (tid == 0) {
        int bi = 0, bv = dist[0];
        for (int r = 1; r < NPTS; ++r) if (dist[r] < bv) { bv = dist[r]; bi = r; }
        best = bi;
    }
    __syncthreads();
    for (int j = tid; j < NPTS; j += blockDim.x) {
        int jj = j - best; if (jj < 0) jj += NPTS;
        firstOut[(b * NPTS + j) * 2 + 0] = fb[jj * 2 + 0];
        firstOut[(b * NPTS + j) * 2 + 1] = fb[jj * 2 + 1];
    }
}

__global__ void init_bnd_kernel(const int* __restrict__ prevB, float* __restrict__ bnd) {
    int i = blockIdx.x * blockDim.x + threadIdx.x;
    if (i < BATCH * NPTS * 2) bnd[i] = (float)prevB[i];
}

__global__ void zero_vt_pad_kernel(__bf16* __restrict__ Vt) {
    long long i = (long long)blockIdx.x * blockDim.x + threadIdx.x;
    if (i >= (long long)BATCH * DPAD * (ATTPAD - NTOK)) return;
    long long b = i / (DPAD * (ATTPAD - NTOK));
    long long rest = i % (DPAD * (ATTPAD - NTOK));
    long long d = rest / (ATTPAD - NTOK);
    long long j = NTOK + rest % (ATTPAD - NTOK);
    Vt[(b * DPAD + d) * ATTPAD + j] = f2bf(0.0f);
}

// ---------------- gather image features + LN0 + positional encoding ----------------
__global__ void gather_ln0_kernel(const float* __restrict__ img,
                                  const int* __restrict__ ibnd,        // mode 0: int boundary
                                  float* __restrict__ bndState,        // mode 1: float boundary (clip+writeback)
                                  int mode, int posBase,
                                  const float* __restrict__ lnw, const float* __restrict__ lnb,
                                  float* __restrict__ Xf, __bf16* __restrict__ Xb) {
    const int i = blockIdx.x, b = blockIdx.y, tid = threadIdx.x;
    __shared__ float tok[DPAD];
    __shared__ float red1[128], red2[128];
    float fy, fx; int y, x;
    if (mode == 0) {
        y = ibnd[(b * NPTS + i) * 2 + 0]; x = ibnd[(b * NPTS + i) * 2 + 1];
        fy = (float)y; fx = (float)x;
    } else {
        fy = bndState[(b * NPTS + i) * 2 + 0];
        fx = bndState[(b * NPTS + i) * 2 + 1];
        fy = fminf(fmaxf(fy, 0.0f), 223.0f);
        fx = fminf(fmaxf(fx, 0.0f), 223.0f);
        y = (int)fy; x = (int)fx;
        if (tid == 0) {
            bndState[(b * NPTS + i) * 2 + 0] = fy;
            bndState[(b * NPTS + i) * 2 + 1] = fx;
        }
    }
    float s1 = 0.0f, s2 = 0.0f;
    for (int d = tid; d < DMODEL; d += blockDim.x) {
        float v;
        if (d < CCH) v = img[((long long)b * CCH + d) * HW + y * WW + x];
        else         v = (d == CCH) ? fy : fx;
        tok[d] = v; s1 += v; s2 += v * v;
    }
    red1[tid] = s1; red2[tid] = s2; __syncthreads();
    for (int s = blockDim.x / 2; s > 0; s >>= 1) {
        if (tid < s) { red1[tid] += red1[tid + s]; red2[tid] += red2[tid + s]; }
        __syncthreads();
    }
    const float mean = red1[0] / (float)DMODEL;
    const float var  = red2[0] / (float)DMODEL - mean * mean;
    const float rstd = rsqrtf(var + 1e-5f);
    const int pos = posBase + i;
    const long long rowOff = ((long long)b * NTOK + pos) * DPAD;
    const float lk = logf(10000.0f) / (float)DMODEL;
    for (int d = tid; d < DPAD; d += blockDim.x) {
        float v = 0.0f;
        if (d < DMODEL) {
            float freq = __expf(-(float)(d & ~1) * lk);
            float pe = (d & 1) ? __cosf((float)pos * freq) : __sinf((float)pos * freq);
            v = (tok[d] - mean) * rstd * lnw[d] + lnb[d] + pe;
        }
        Xf[rowOff + d] = v;
        Xb[rowOff + d] = f2bf(v);
    }
}

// ---------------- residual + LayerNorm ----------------
__global__ void ln_res_kernel(const float* __restrict__ A, const float* __restrict__ Bv,
                              int inRowBase,
                              const float* __restrict__ lnw, const float* __restrict__ lnb,
                              float* __restrict__ outF, long long outBStride,
                              __bf16* __restrict__ outB) {
    const int rr = blockIdx.x, b = blockIdx.y, tid = threadIdx.x;
    const long long inOff = ((long long)b * NTOK + inRowBase + rr) * DPAD;
    __shared__ float tok[DPAD];
    __shared__ float red1[128], red2[128];
    float s1 = 0.0f, s2 = 0.0f;
    for (int d = tid; d < DMODEL; d += blockDim.x) {
        float v = A[inOff + d] + Bv[inOff + d];
        tok[d] = v; s1 += v; s2 += v * v;
    }
    red1[tid] = s1; red2[tid] = s2; __syncthreads();
    for (int s = blockDim.x / 2; s > 0; s >>= 1) {
        if (tid < s) { red1[tid] += red1[tid + s]; red2[tid] += red2[tid + s]; }
        __syncthreads();
    }
    const float mean = red1[0] / (float)DMODEL;
    const float var  = red2[0] / (float)DMODEL - mean * mean;
    const float rstd = rsqrtf(var + 1e-5f);
    const long long outOff = (long long)b * outBStride + (long long)rr * DPAD;
    for (int d = tid; d < DPAD; d += blockDim.x) {
        float v = (d < DMODEL) ? (tok[d] - mean) * rstd * lnw[d] + lnb[d] : 0.0f;
        outF[outOff + d] = v;
        if (outB) outB[outOff + d] = f2bf(v);
    }
}

// ---------------- row softmax over 240 cols (writes bf16, zero pad to 256) ----------------
__global__ void softmax_kernel(const float* __restrict__ Sf, __bf16* __restrict__ Ab) {
    const int i = blockIdx.x, b = blockIdx.y, tid = threadIdx.x;
    const float* row = Sf + ((long long)b * NTOK + i) * ATTPAD;
    __shared__ float red[256];
    float m = -1e30f;
    for (int j = tid; j < NTOK; j += blockDim.x) m = fmaxf(m, row[j]);
    red[tid] = m; __syncthreads();
    for (int s = blockDim.x / 2; s > 0; s >>= 1) {
        if (tid < s) red[tid] = fmaxf(red[tid], red[tid + s]);
        __syncthreads();
    }
    m = red[0]; __syncthreads();
    float sum = 0.0f;
    for (int j = tid; j < NTOK; j += blockDim.x) sum += __expf(row[j] - m);
    red[tid] = sum; __syncthreads();
    for (int s = blockDim.x / 2; s > 0; s >>= 1) {
        if (tid < s) red[tid] += red[tid + s];
        __syncthreads();
    }
    const float inv = 1.0f / red[0];
    __bf16* arow = Ab + ((long long)b * NTOK + i) * ATTPAD;
    for (int j = tid; j < ATTPAD; j += blockDim.x) {
        float v = (j < NTOK) ? __expf(row[j] - m) * inv : 0.0f;
        arow[j] = f2bf(v);
    }
}

// ---------------- final head: off = out[80:160] @ w_fc^T + b_fc ; boundary update ----------------
__global__ void head_kernel(const float* __restrict__ O2, const float* __restrict__ wfc,
                            const float* __restrict__ bfc,
                            float* __restrict__ bndState, float* __restrict__ outRes) {
    const int i = blockIdx.x, b = blockIdx.y, tid = threadIdx.x;
    const float* row = O2 + ((long long)b * NPTS + i) * DPAD;
    __shared__ float r0[128], r1[128];
    float s0 = 0.0f, s1 = 0.0f;
    for (int d = tid; d < DMODEL; d += blockDim.x) {
        float v = row[d];
        s0 += v * wfc[d];
        s1 += v * wfc[DMODEL + d];
    }
    r0[tid] = s0; r1[tid] = s1; __syncthreads();
    for (int s = blockDim.x / 2; s > 0; s >>= 1) {
        if (tid < s) { r0[tid] += r0[tid + s]; r1[tid] += r1[tid + s]; }
        __syncthreads();
    }
    if (tid == 0) {
        float n0 = bndState[(b * NPTS + i) * 2 + 0] + r0[0] + bfc[0];
        float n1 = bndState[(b * NPTS + i) * 2 + 1] + r1[0] + bfc[1];
        bndState[(b * NPTS + i) * 2 + 0] = n0;
        bndState[(b * NPTS + i) * 2 + 1] = n1;
        outRes[(b * NPTS + i) * 2 + 0] = n0;
        outRes[(b * NPTS + i) * 2 + 1] = n1;
    }
}

// =======================================================================================
extern "C" void kernel_launch(void* const* d_in, const int* in_sizes, int n_in,
                              void* d_out, int out_size, void* d_ws, size_t ws_size,
                              hipStream_t stream) {
    const float* pre_img   = (const float*)d_in[0];
    const float* cur_img   = (const float*)d_in[1];
    const float* first_img = (const float*)d_in[2];
    const int*   first_bnd = (const int*)d_in[3];
    const int*   prev_bnd  = (const int*)d_in[4];
    const float* ln0w = (const float*)d_in[5],  *ln0b = (const float*)d_in[6];
    const float* wq   = (const float*)d_in[7],  *bq   = (const float*)d_in[8];
    const float* wk   = (const float*)d_in[9],  *bk   = (const float*)d_in[10];
    const float* wv   = (const float*)d_in[11], *bv   = (const float*)d_in[12];
    const float* wo   = (const float*)d_in[13], *bo   = (const float*)d_in[14];
    const float* ln1w = (const float*)d_in[15], *ln1b = (const float*)d_in[16];
    const float* wff1 = (const float*)d_in[17], *bff1 = (const float*)d_in[18];
    const float* wff2 = (const float*)d_in[19], *bff2 = (const float*)d_in[20];
    const float* ln2w = (const float*)d_in[21], *ln2b = (const float*)d_in[22];
    const float* wfc  = (const float*)d_in[23], *bfc  = (const float*)d_in[24];

    char* ws = (char*)d_ws;
    size_t off = 0;
    auto alloc = [&](size_t bytes) -> char* {
        char* p = ws + off;
        off = (off + bytes + 255) & ~(size_t)255;
        return p;
    };
    // bf16 padded weights
    __bf16* wq_bf  = (__bf16*)alloc((size_t)DPAD * DPAD * 2);
    __bf16* wk_bf  = (__bf16*)alloc((size_t)DPAD * DPAD * 2);
    __bf16* wv_bf  = (__bf16*)alloc((size_t)DPAD * DPAD * 2);
    __bf16* wo_bf  = (__bf16*)alloc((size_t)DPAD * DPAD * 2);
    __bf16* ff1_bf = (__bf16*)alloc((size_t)FFDIM * DPAD * 2);
    __bf16* ff2_bf = (__bf16*)alloc((size_t)DPAD * FFDIM * 2);
    // state + activations
    int*    firstB = (int*)   alloc((size_t)BATCH * NPTS * 2 * 4);
    float*  bnd    = (float*) alloc((size_t)BATCH * NPTS * 2 * 4);
    float*  Xf     = (float*) alloc((size_t)BATCH * NTOK * DPAD * 4);
    __bf16* Xb     = (__bf16*)alloc((size_t)BATCH * NTOK * DPAD * 2);
    __bf16* Qb     = (__bf16*)alloc((size_t)BATCH * NTOK * DPAD * 2);
    __bf16* Kb     = (__bf16*)alloc((size_t)BATCH * NTOK * DPAD * 2);
    __bf16* Vt     = (__bf16*)alloc((size_t)BATCH * DPAD * ATTPAD * 2);
    float*  Sf     = (float*) alloc((size_t)BATCH * NTOK * ATTPAD * 4);
    __bf16* Ab     = (__bf16*)alloc((size_t)BATCH * NTOK * ATTPAD * 2);
    __bf16* AVb    = (__bf16*)alloc((size_t)BATCH * NTOK * DPAD * 2);
    float*  AVO    = (float*) alloc((size_t)BATCH * NTOK * DPAD * 4);
    float*  Hf     = (float*) alloc((size_t)BATCH * NTOK * DPAD * 4);
    __bf16* Hb     = (__bf16*)alloc((size_t)BATCH * NTOK * DPAD * 2);
    __bf16* Tb     = (__bf16*)alloc((size_t)BATCH * NTOK * FFDIM * 2);
    float*  Y2     = (float*) alloc((size_t)BATCH * NTOK * DPAD * 4);
    float*  O2     = (float*) alloc((size_t)BATCH * NPTS * DPAD * 4);

    auto padw = [&](const float* src, __bf16* dst, int rows, int cols, int npad, int kpad) {
        long long total = (long long)npad * kpad;
        pad_w_kernel<<<dim3((unsigned)((total + 255) / 256)), 256, 0, stream>>>(src, dst, rows, cols, kpad, total);
    };
    padw(wq,   wq_bf,  DMODEL, DMODEL, DPAD,  DPAD);
    padw(wk,   wk_bf,  DMODEL, DMODEL, DPAD,  DPAD);
    padw(wv,   wv_bf,  DMODEL, DMODEL, DPAD,  DPAD);
    padw(wo,   wo_bf,  DMODEL, DMODEL, DPAD,  DPAD);
    padw(wff1, ff1_bf, FFDIM,  DMODEL, FFDIM, DPAD);
    padw(wff2, ff2_bf, DMODEL, FFDIM,  DPAD,  FFDIM);

    align_kernel<<<dim3(BATCH), 128, 0, stream>>>(first_bnd, prev_bnd, firstB);
    init_bnd_kernel<<<dim3((BATCH * NPTS * 2 + 255) / 256), 256, 0, stream>>>(prev_bnd, bnd);
    {
        long long tot = (long long)BATCH * DPAD * (ATTPAD - NTOK);
        zero_vt_pad_kernel<<<dim3((unsigned)((tot + 255) / 256)), 256, 0, stream>>>(Vt);
    }

    // static token rows (constant across refine iterations)
    gather_ln0_kernel<<<dim3(NPTS, BATCH), 128, 0, stream>>>(pre_img,   prev_bnd, nullptr, 0, 0,   ln0w, ln0b, Xf, Xb);
    gather_ln0_kernel<<<dim3(NPTS, BATCH), 128, 0, stream>>>(first_img, firstB,   nullptr, 0, 160, ln0w, ln0b, Xf, Xb);

    const long long sTokD  = (long long)NTOK * DPAD;
    const long long sTokA  = (long long)NTOK * ATTPAD;
    const long long sVt    = (long long)DPAD * ATTPAD;
    const long long sTokFF = (long long)NTOK * FFDIM;
    const float qscale = rsqrtf((float)DMODEL);

    auto gemm = [&](GemmArgs g, int ntiles) {
        gemm_bf16_wmma<<<dim3(ntiles, NTOK / 16, BATCH), 32, 0, stream>>>(g);
    };

    for (int it = 0; it < 3; ++it) {
        // current-frame tokens (clip boundary, gather, LN0+pe)
        gather_ln0_kernel<<<dim3(NPTS, BATCH), 128, 0, stream>>>(cur_img, nullptr, bnd, 1, 80, ln0w, ln0b, Xf, Xb);

        // Q = (X wq^T + bq) * scale   (bf16 out)
        gemm({Xb, DPAD, sTokD, wq_bf, DPAD, 0, bq, DMODEL, qscale, 0,
              nullptr, 0, 0, Qb, DPAD, sTokD, 0, DPAD / 32}, DPAD / 16);
        // K = X wk^T + bk
        gemm({Xb, DPAD, sTokD, wk_bf, DPAD, 0, bk, DMODEL, 1.0f, 0,
              nullptr, 0, 0, Kb, DPAD, sTokD, 0, DPAD / 32}, DPAD / 16);
        // V = X wv^T + bv  (stored transposed d-major for the A*V GEMM)
        gemm({Xb, DPAD, sTokD, wv_bf, DPAD, 0, bv, DMODEL, 1.0f, 0,
              nullptr, 0, 0, Vt, ATTPAD, sVt, 1, DPAD / 32}, DPAD / 16);
        // S = Q K^T  (f32 out, 240x240)
        gemm({Qb, DPAD, sTokD, Kb, DPAD, sTokD, nullptr, 0, 1.0f, 0,
              Sf, ATTPAD, sTokA, nullptr, 0, 0, 0, DPAD / 32}, NTOK / 16);
        // softmax rows -> bf16 attention matrix (zero-padded to 256)
        softmax_kernel<<<dim3(NTOK, BATCH), 256, 0, stream>>>(Sf, Ab);
        // AV = A V   (bf16 out)
        gemm({Ab, ATTPAD, sTokA, Vt, ATTPAD, sVt, nullptr, 0, 1.0f, 0,
              nullptr, 0, 0, AVb, DPAD, sTokD, 0, ATTPAD / 32}, DPAD / 16);
        // AVO = AV wo^T + bo  (f32 out)
        gemm({AVb, DPAD, sTokD, wo_bf, DPAD, 0, bo, DMODEL, 1.0f, 0,
              AVO, DPAD, sTokD, nullptr, 0, 0, 0, DPAD / 32}, DPAD / 16);
        // h = LN1(x + AVO)
        ln_res_kernel<<<dim3(NTOK, BATCH), 128, 0, stream>>>(Xf, AVO, 0, ln1w, ln1b, Hf, sTokD, Hb);
        // T = relu(h ff1^T + b_ff1)  (bf16 out)
        gemm({Hb, DPAD, sTokD, ff1_bf, DPAD, 0, bff1, FFDIM, 1.0f, 1,
              nullptr, 0, 0, Tb, FFDIM, sTokFF, 0, DPAD / 32}, FFDIM / 16);
        // Y2 = T ff2^T + b_ff2  (f32 out)
        gemm({Tb, FFDIM, sTokFF, ff2_bf, FFDIM, 0, bff2, DMODEL, 1.0f, 0,
              Y2, DPAD, sTokD, nullptr, 0, 0, 0, FFDIM / 32}, DPAD / 16);
        // out rows 80..159 = LN2(h + Y2)   (only the head rows are needed)
        ln_res_kernel<<<dim3(NPTS, BATCH), 128, 0, stream>>>(Hf, Y2, 80, ln2w, ln2b, O2, (long long)NPTS * DPAD, nullptr);
        // boundary update + result slice
        head_kernel<<<dim3(NPTS, BATCH), 128, 0, stream>>>(O2, wfc, bfc, bnd,
                                                           (float*)d_out + (size_t)it * BATCH * NPTS * 2);
    }
    (void)in_sizes; (void)n_in; (void)out_size; (void)ws_size;
}
